// Attention_60842506715595
// MI455X (gfx1250) — matrix operand order repaired and checked by reading, not compiled, not run on hardware
//
#include <hip/hip_runtime.h>
#include <hip/hip_bf16.h>

typedef __bf16 bf16_t;
typedef __attribute__((ext_vector_type(16))) __bf16 bf16x16;
typedef __attribute__((ext_vector_type(8)))  float  f32x8;
typedef __attribute__((ext_vector_type(4))) unsigned int u32x4;
typedef __attribute__((ext_vector_type(8))) int i32x8;
typedef __attribute__((ext_vector_type(4))) int i32x4;

#define DIMV   4096
#define NH     32
#define NKV    8
#define HDIM   128
#define BATCH  2
#define SEQ    2048
#define WIN    1024
#define TOK    (BATCH*SEQ)

// ---------- fragment load helper: two 16B chunks -> 16 x bf16 ----------
__device__ __forceinline__ bf16x16 load16(const bf16_t* p0, const bf16_t* p1) {
    union { bf16x16 v; uint4 u[2]; } r;
    r.u[0] = *(const uint4*)p0;
    r.u[1] = *(const uint4*)p1;
    return r.v;
}

__device__ __forceinline__ f32x8 wmma_bf16(bf16x16 a, bf16x16 b, f32x8 c) {
    return __builtin_amdgcn_wmma_f32_16x16x32_bf16(false, a, false, b,
                                                   (short)0, c, false, false);
}

// ---------- TDM: DMA a (tileD1 rows x 32 bf16) tile from global to LDS ----------
// D# packing per CDNA5 ISA (2-D tensor, groups 2/3 zero).
__device__ __forceinline__ void tdm_load_tile(const bf16_t* gsrc, int elemStride,
                                              uint32_t ldsOff) {
    uint64_t ga = (uint64_t)(uintptr_t)gsrc;
    u32x4 g0;
    g0[0] = 1u;                                    // count=1, user descriptor
    g0[1] = ldsOff;                                // lds_addr (bytes)
    g0[2] = (uint32_t)ga;                          // global_addr[31:0]
    g0[3] = (uint32_t)(ga >> 32) | (2u << 30);     // global_addr[56:32] | type=2
    i32x8 g1;
    const uint32_t TD0 = 0x7fffffffu, TD1 = 0x7fffffffu;   // tensor dims: huge (tile never OOB)
    g1[0] = (int)0x00010000u;                      // wg_mask=0, data_size=1 (2 bytes)
    g1[1] = (int)((TD0 & 0xffffu) << 16);          // [31:16]=tensor_dim0[15:0]
    g1[2] = (int)(((TD0 >> 16) & 0xffffu) | ((TD1 & 0xffffu) << 16));
    g1[3] = (int)(((TD1 >> 16) & 0xffffu) | (32u << 16));  // tile_dim0 = 32 elems
    g1[4] = (int)128u;                             // tile_dim1 = 128 rows, tile_dim2 = 0
    g1[5] = (int)(uint32_t)elemStride;             // tensor_dim0_stride[31:0] (elements)
    g1[6] = 0;                                     // stride hi, dim1_stride lo
    g1[7] = 0;                                     // dim1_stride hi
    i32x4 z4 = {0, 0, 0, 0};
#if __clang_major__ >= 23
    i32x8 z8 = {0, 0, 0, 0, 0, 0, 0, 0};
    __builtin_amdgcn_tensor_load_to_lds(g0, g1, z4, z4, z8, 0);
#else
    __builtin_amdgcn_tensor_load_to_lds(g0, g1, z4, z4, 0);
#endif
}

// ---------- elementwise f32 -> bf16 ----------
__global__ void f32_to_bf16_k(const float* __restrict__ in, bf16_t* __restrict__ out, int n) {
    int i = blockIdx.x * blockDim.x + threadIdx.x;
    if (i < n) out[i] = (bf16_t)in[i];
}

// ---------- W (K x N) f32 -> Wt (N x K) bf16, 32x32 LDS tiles ----------
__global__ __launch_bounds__(256)
void transpose_bf16_k(const float* __restrict__ W, bf16_t* __restrict__ Wt, int K, int N) {
    __shared__ bf16_t tile[32][33];
    int k0 = blockIdx.y * 32, n0 = blockIdx.x * 32;
    int tx = threadIdx.x & 31, ty = threadIdx.x >> 5;   // 8 rows of 32
#pragma unroll
    for (int i = 0; i < 32; i += 8)
        tile[ty + i][tx] = (bf16_t)W[(size_t)(k0 + ty + i) * N + n0 + tx];
    __syncthreads();
#pragma unroll
    for (int i = 0; i < 32; i += 8)
        Wt[(size_t)(n0 + ty + i) * K + k0 + tx] = tile[tx][ty + i];
}

// ---------- bf16 GEMM:  C(MxN) = A(MxK,row) * Bt(NxK,row)^T ----------
// Block = 8 waves -> 256x128 tile.  B tile (128x32, 8KB) is TDM-DMA'd to LDS
// (double buffered, TENSORcnt + barrier pipelined); A streams via VMEM + prefetch.
// MODE 0: f32 row-major store
// MODE 1: RoPE (cos/sin per token m, pair (n%HDIM)) then *oscale, bf16 row-major
// MODE 2: bf16 transposed store:  C[b][n][s]  (b=m/seq, s=m%seq)
template<int MODE>
__global__ __launch_bounds__(256)
void gemm_bf16_k(const bf16_t* __restrict__ A, const bf16_t* __restrict__ Bt,
                 void* __restrict__ Cout, int M, int N, int K,
                 const float* __restrict__ cosT, const float* __restrict__ sinT,
                 float oscale, int seq)
{
    __shared__ bf16_t ldsB[2][128 * 32];                  // 2 x 8 KB
    const int lane = threadIdx.x & 31;
    const int wid  = threadIdx.x >> 5;                    // 0..7
    const int m0 = blockIdx.y * 256 + (wid >> 1) * 64;
    const int n0 = blockIdx.x * 128 + (wid & 1) * 64;
    const int nblk = blockIdx.x * 128;                    // block's B row base
    const int g  = lane >> 4;
    const int lm = lane & 15;

    f32x8 acc[4][4] = {};

    // prologue: DMA first B tile
    if (wid == 0) {
        tdm_load_tile(Bt + (size_t)nblk * K, K, (uint32_t)(uintptr_t)&ldsB[0][0]);
        __builtin_amdgcn_s_wait_tensorcnt(0);
    }
    __syncthreads();

    for (int k = 0; k < K; k += 32) {
        const int cur = (k >> 5) & 1;
        // kick off DMA of next B tile into the other buffer
        if (wid == 0 && k + 32 < K)
            tdm_load_tile(Bt + (size_t)nblk * K + (k + 32), K,
                          (uint32_t)(uintptr_t)&ldsB[cur ^ 1][0]);

        bf16x16 af[4], bfr[4];
#pragma unroll
        for (int mt = 0; mt < 4; ++mt) {
            const bf16_t* pa = A + (size_t)(m0 + mt * 16 + lm) * K + k + g * 8;
            af[mt] = load16(pa, pa + 16);                 // K: g*8..+7 and 16+g*8..+7
            __builtin_prefetch(pa + 128, 0, 1);           // global_prefetch_b8, 4 steps ahead
        }
#pragma unroll
        for (int nt = 0; nt < 4; ++nt) {
            const bf16_t* pb = &ldsB[cur][((wid & 1) * 64 + nt * 16 + lm) * 32 + g * 8];
            bfr[nt] = load16(pb, pb + 16);                // ds_load_b128 x2
        }
#pragma unroll
        for (int mt = 0; mt < 4; ++mt)
#pragma unroll
            for (int nt = 0; nt < 4; ++nt)
                acc[mt][nt] = wmma_bf16(af[mt], bfr[nt], acc[mt][nt]);

        if (wid == 0)
            __builtin_amdgcn_s_wait_tensorcnt(0);         // next tile landed
        __syncthreads();                                   // publish buffer / allow reuse
    }

#pragma unroll
    for (int mt = 0; mt < 4; ++mt)
#pragma unroll
        for (int nt = 0; nt < 4; ++nt)
#pragma unroll
            for (int r = 0; r < 8; ++r) {
                int m = m0 + mt * 16 + r + g * 8;
                int n = n0 + nt * 16 + lm;
                float v = acc[mt][nt][r];
                if (MODE == 0) {
                    ((float*)Cout)[(size_t)m * N + n] = v;
                } else if (MODE == 1) {
                    int i = (n & (HDIM - 1)) >> 1;
                    float c = cosT[(size_t)m * (HDIM / 2) + i];
                    float s = sinT[(size_t)m * (HDIM / 2) + i];
                    float p = __shfl_xor(v, 1, 32);       // partner of (2i,2i+1) pair
                    float o = (n & 1) ? fmaf(p, s, v * c) : fmaf(v, c, -(p * s));
                    ((bf16_t*)Cout)[(size_t)m * N + n] = (bf16_t)(o * oscale);
                } else {
                    ((bf16_t*)Cout)[(size_t)(m / seq) * N * seq + (size_t)n * seq + (m % seq)] = (bf16_t)v;
                }
            }
}

// ---------- flash attention: 1 wave = 16 q rows of one (b,h) ----------
__global__ __launch_bounds__(128)
void attn_k(const bf16_t* __restrict__ Q,   // (T, NH*HDIM), pre-scaled by 1/sqrt(HDIM)
            const bf16_t* __restrict__ Kc,  // (T, NKV*HDIM)
            const bf16_t* __restrict__ Vt,  // (B, NKV*HDIM, SEQ)
            bf16_t* __restrict__ O)         // (T, NH*HDIM)
{
    __shared__ bf16_t ldsP[4][16 * 32];
    const int lane = threadIdx.x & 31;
    const int wid  = threadIdx.x >> 5;
    const int qt = blockIdx.x * 4 + wid;      // q tile (16 rows), 0..SEQ/16-1
    const int h  = blockIdx.y;
    const int b  = blockIdx.z;
    const int kv = h >> 2;                    // H/KV = 4
    const int g  = lane >> 4;
    const int lm = lane & 15;

    // Q fragments: 16x128 = 4 A-frags of 16x32
    bf16x16 qf[4];
    const bf16_t* qbase = Q + ((size_t)(b * SEQ + qt * 16 + lm) * NH + h) * HDIM;
#pragma unroll
    for (int kc = 0; kc < 4; ++kc) {
        const bf16_t* p = qbase + kc * 32 + g * 8;
        qf[kc] = load16(p, p + 16);
    }

    f32x8 oac[8] = {};
    float mrow[8], lrow[8];
#pragma unroll
    for (int r = 0; r < 8; ++r) { mrow[r] = -1e30f; lrow[r] = 0.f; }

    const int qlo = qt * 16, qhi = qt * 16 + 15;
    int kb0 = qlo - (WIN - 1); if (kb0 < 0) kb0 = 0; kb0 &= ~31;
    const int kb1 = qhi & ~31;                // start of last 32-key block

    for (int kb = kb0; kb <= kb1; kb += 32) {
        // ---- S = Q * K^T : two 16x16 score tiles (keys kb..+15, kb+16..+31)
        f32x8 sc[2] = {};
#pragma unroll
        for (int t = 0; t < 2; ++t)
#pragma unroll
            for (int kc = 0; kc < 4; ++kc) {
                const bf16_t* p = Kc + ((size_t)(b * SEQ + kb + t * 16 + lm) * NKV + kv) * HDIM
                                  + kc * 32 + g * 8;
                sc[t] = wmma_bf16(qf[kc], load16(p, p + 16), sc[t]);
            }

        // ---- online softmax (row r lives in lane-group g, lanes lm=0..15)
        float pscale[8];
#pragma unroll
        for (int r = 0; r < 8; ++r) {
            int row  = qt * 16 + r + g * 8;
            int key0 = kb + lm, key1 = kb + 16 + lm;
            float s0 = sc[0][r], s1 = sc[1][r];
            if (key0 > row || row - key0 >= WIN) s0 = -1e30f;
            if (key1 > row || row - key1 >= WIN) s1 = -1e30f;
            float t = fmaxf(s0, s1);
            t = fmaxf(t, __shfl_xor(t, 1, 32));
            t = fmaxf(t, __shfl_xor(t, 2, 32));
            t = fmaxf(t, __shfl_xor(t, 4, 32));
            t = fmaxf(t, __shfl_xor(t, 8, 32));
            float mnew = fmaxf(mrow[r], t);
            float scl  = __expf(mrow[r] - mnew);
            float p0 = __expf(s0 - mnew);
            float p1 = __expf(s1 - mnew);
            float rs = p0 + p1;
            rs += __shfl_xor(rs, 1, 32);
            rs += __shfl_xor(rs, 2, 32);
            rs += __shfl_xor(rs, 4, 32);
            rs += __shfl_xor(rs, 8, 32);
            lrow[r] = lrow[r] * scl + rs;
            mrow[r] = mnew;
            pscale[r] = scl;
            ldsP[wid][(r + g * 8) * 32 + lm]      = (bf16_t)p0;   // P tile, row-major 16x32
            ldsP[wid][(r + g * 8) * 32 + 16 + lm] = (bf16_t)p1;
        }
        asm volatile("s_wait_dscnt 0x0" ::: "memory");   // LDS RAW within wave

        // P as A-fragment (16x32)
        const bf16_t* pp = &ldsP[wid][lm * 32 + g * 8];
        bf16x16 pf = load16(pp, pp + 16);

        // rescale O, then O += P * V
#pragma unroll
        for (int nt = 0; nt < 8; ++nt)
#pragma unroll
            for (int r = 0; r < 8; ++r) oac[nt][r] *= pscale[r];
#pragma unroll
        for (int nt = 0; nt < 8; ++nt) {
            const bf16_t* pv = Vt + ((size_t)(b * NKV + kv) * HDIM + nt * 16 + lm) * SEQ
                               + kb + g * 8;
            oac[nt] = wmma_bf16(pf, load16(pv, pv + 16), oac[nt]);
        }
    }

    float inv[8];
#pragma unroll
    for (int r = 0; r < 8; ++r) inv[r] = 1.0f / lrow[r];
#pragma unroll
    for (int nt = 0; nt < 8; ++nt)
#pragma unroll
        for (int r = 0; r < 8; ++r) {
            int row = qt * 16 + r + g * 8;
            int hd  = nt * 16 + lm;
            O[((size_t)(b * SEQ + row) * NH + h) * HDIM + hd] = (bf16_t)(oac[nt][r] * inv[r]);
        }
}

// ------------------------------------------------------------------
extern "C" void kernel_launch(void* const* d_in, const int* in_sizes, int n_in,
                              void* d_out, int out_size, void* d_ws, size_t ws_size,
                              hipStream_t stream) {
    (void)in_sizes; (void)n_in; (void)out_size; (void)ws_size;
    const float* x    = (const float*)d_in[0];
    const float* cosT = (const float*)d_in[1];
    const float* sinT = (const float*)d_in[2];
    const float* wq   = (const float*)d_in[3];
    const float* wk   = (const float*)d_in[4];
    const float* wv   = (const float*)d_in[5];
    const float* wo   = (const float*)d_in[6];

    char* w = (char*)d_ws;
    bf16_t* xb = (bf16_t*)(w);                          // 32 MB  x in bf16
    bf16_t* wT = (bf16_t*)(w + ((size_t)32 << 20));     // 32 MB  current weight^T (reused)
    bf16_t* Qb = (bf16_t*)(w + ((size_t)64 << 20));     // 32 MB  Q (roped, scaled)
    bf16_t* Kb = (bf16_t*)(w + ((size_t)96 << 20));     //  8 MB  K (roped)
    bf16_t* Vt = (bf16_t*)(w + ((size_t)104 << 20));    //  8 MB  V transposed
    bf16_t* Ab = (bf16_t*)(w + ((size_t)112 << 20));    // 32 MB  attention output bf16

    const float qscale = 0.08838834764831845f;          // 1/sqrt(128)

    // x -> bf16
    {
        int n = TOK * DIMV;
        f32_to_bf16_k<<<(n + 255) / 256, 256, 0, stream>>>(x, xb, n);
    }

    // Q = rope(x @ wq) * qscale
    transpose_bf16_k<<<dim3(DIMV / 32, DIMV / 32), 256, 0, stream>>>(wq, wT, DIMV, NH * HDIM);
    gemm_bf16_k<1><<<dim3((NH * HDIM) / 128, TOK / 256), 256, 0, stream>>>(
        xb, wT, Qb, TOK, NH * HDIM, DIMV, cosT, sinT, qscale, SEQ);

    // K = rope(x @ wk)
    transpose_bf16_k<<<dim3((NKV * HDIM) / 32, DIMV / 32), 256, 0, stream>>>(wk, wT, DIMV, NKV * HDIM);
    gemm_bf16_k<1><<<dim3((NKV * HDIM) / 128, TOK / 256), 256, 0, stream>>>(
        xb, wT, Kb, TOK, NKV * HDIM, DIMV, cosT, sinT, 1.0f, SEQ);

    // Vt = (x @ wv) transposed to (B, NKV*HDIM, SEQ)
    transpose_bf16_k<<<dim3((NKV * HDIM) / 32, DIMV / 32), 256, 0, stream>>>(wv, wT, DIMV, NKV * HDIM);
    gemm_bf16_k<2><<<dim3((NKV * HDIM) / 128, TOK / 256), 256, 0, stream>>>(
        xb, wT, Vt, TOK, NKV * HDIM, DIMV, cosT, sinT, 1.0f, SEQ);

    // attention
    attn_k<<<dim3(SEQ / 64, NH, BATCH), 128, 0, stream>>>(Qb, Kb, Vt, Ab);

    // out = attn @ wo  (f32)
    transpose_bf16_k<<<dim3(DIMV / 32, (NH * HDIM) / 32), 256, 0, stream>>>(wo, wT, NH * HDIM, DIMV);
    gemm_bf16_k<0><<<dim3(DIMV / 128, TOK / 256), 256, 0, stream>>>(
        Ab, wT, (float*)d_out, TOK, DIMV, NH * HDIM, cosT, sinT, 1.0f, SEQ);
}
